// FixedDepthRnnFlexibleNonlinearity_17231408791954
// MI455X (gfx1250) — matrix operand order repaired and checked by reading, not compiled, not run on hardware
//
#include <hip/hip_runtime.h>
#include <hip/hip_bf16.h>

typedef __attribute__((ext_vector_type(16))) __bf16 bf16x16;
typedef __attribute__((ext_vector_type(8)))  __bf16 bf16x8;
typedef __attribute__((ext_vector_type(8)))  float  f32x8;

#define B_ 128
#define T_ 512
#define I_ 64
#define R_ 512
#define O_ 64

#define LDX (R_ + 8)   // padded LDS stride for hidden state (bf16 elems)
#define LDP (I_ + 8)   // padded LDS stride for x_pred tile

// ---------------------------------------------------------------------------
// Packed-fragment weight layout (B-operand order for v_wmma_f32_16x16x32_bf16):
//   block bi = ct*KC + kc   (ct = 16-column tile, kc = 32-wide K chunk)
//   within block: lane l (0..31) holds 16 contiguous bf16:
//       W[ct*16 + (l&15)][kc*32 + 8*(l>>4) + (i&7) + 16*(i>>3)],  i = 0..15
// One fragment = 1 KB, lane-contiguous -> perfectly coalesced b128 loads.
// ws layout: Whp [R*R] ; Uhp [R*I] ; Woutp [O*R]   (bf16, 640 KB total)
// ---------------------------------------------------------------------------
__device__ __forceinline__ void pack_one(const float* __restrict__ W,
                                         __bf16* __restrict__ dst,
                                         int idx, int KK, int KC) {
    int i  = idx & 15;
    int l  = (idx >> 4) & 31;
    int bi = idx >> 9;
    int ct = bi / KC;
    int kc = bi - ct * KC;
    int col = ct * 16 + (l & 15);
    int k   = kc * 32 + 8 * (l >> 4) + (i & 7) + 16 * (i >> 3);
    dst[idx] = (__bf16)W[(size_t)col * KK + k];
}

__global__ void pack_weights_kernel(const float* __restrict__ Wh,
                                    const float* __restrict__ Uh,
                                    const float* __restrict__ Wout,
                                    __bf16* __restrict__ ws) {
    int idx = blockIdx.x * blockDim.x + threadIdx.x;
    const int nWh = R_ * R_;
    const int nUh = R_ * I_;
    const int nWo = O_ * R_;
    if (idx < nWh) {
        pack_one(Wh, ws, idx, R_, R_ / 32);
    } else if (idx < nWh + nUh) {
        pack_one(Uh, ws + nWh, idx - nWh, I_, I_ / 32);
    } else if (idx < nWh + nUh + nWo) {
        pack_one(Wout, ws + nWh + nUh, idx - nWh - nUh, R_, R_ / 32);
    }
}

// Fragment load from packed weights (global or LDS): 32B lane-contiguous.
__device__ __forceinline__ bf16x16 ldfrag(const __bf16* base, int ct, int KC,
                                          int kc, int lane) {
    return *(const bf16x16*)(base + (((size_t)(ct * KC + kc) * 32 + lane) << 4));
}

// A-fragment (16x32 tile row) from LDS, per ISA layout:
// elements 0-7 = K kbase..kbase+7, elements 8-15 = K kbase+16..kbase+23,
// kbase = kc*32 + 8*half.
__device__ __forceinline__ bf16x16 load_afrag(const __bf16* X, int ldx, int m, int kbase) {
    const __bf16* p = X + m * ldx + kbase;
    bf16x8 lo = *(const bf16x8*)(p);
    bf16x8 hi = *(const bf16x8*)(p + 16);
    bf16x16 a;
#pragma unroll
    for (int i = 0; i < 8; ++i) { a[i] = lo[i]; a[i + 8] = hi[i]; }
    return a;
}

#define WMMA_BF16(A, Bf, C) \
    __builtin_amdgcn_wmma_f32_16x16x32_bf16(false, (A), false, (Bf), (short)0, (C), false, false)

// ---------------------------------------------------------------------------
// Persistent per-batch-tile RNN scan.
// grid = 8 blocks (16 batch rows each), 256 threads = 8 waves.
// Wave w owns hidden columns [w*64, w*64+64) as 4 N-tiles.
// VGPR-resident: W_h frags for K=0..127 (BWh[4][4], 128 VGPRs).
// LDS-resident:  U_h (64 KB packed) + hidden state + x_pred tile.
// L2-streamed:   W_h K=128..511, W_out (packed, coalesced) — opaque offsets
//                defeat LICM so these stay as in-loop global_load_b128.
// ---------------------------------------------------------------------------
__global__ void __launch_bounds__(256, 1)
rnn_scan_kernel(const float* __restrict__ x_pred,   // [B,T,I]
                const float* __restrict__ h0,       // [B,R]
                const float* __restrict__ b_h,      // [R]
                const float* __restrict__ b_out,    // [O]
                const __bf16* __restrict__ wbf,     // packed weights
                float* __restrict__ out)            // y[B,T,O] ++ xlast ++ xlast
{
    const __bf16* Whp   = wbf;
    const __bf16* Uhp   = wbf + (size_t)R_ * R_;
    const __bf16* Woutp = Uhp + (size_t)R_ * I_;

    __shared__ __align__(16) __bf16 Xbf[16][LDX];
    __shared__ __align__(16) __bf16 XPbf[16][LDP];
    __shared__ __align__(16) __bf16 UhL[R_ * I_];   // packed U_h, 64 KB

    const int tid  = threadIdx.x;
    const int wave = tid >> 5;
    const int lane = tid & 31;
    const int half = lane >> 4;
    const int lm   = lane & 15;
    const int b0   = blockIdx.x * 16;

    // Per-lane biases for the 4 owned N-tiles; y-bias for waves 0-3.
    float bhv[4];
#pragma unroll
    for (int nt = 0; nt < 4; ++nt) bhv[nt] = b_h[wave * 64 + nt * 16 + lm];
    const float bov = (wave < 4) ? b_out[wave * 16 + lm] : 0.0f;

    // ---- VGPR-resident W_h fragments, K = 0..127 (128 VGPRs) ----
    bf16x16 BWh[4][4];
#pragma unroll
    for (int nt = 0; nt < 4; ++nt)
#pragma unroll
        for (int kc = 0; kc < 4; ++kc)
            BWh[nt][kc] = ldfrag(Whp, wave * 4 + nt, 16, kc, lane);

    // ---- copy packed U_h into LDS (linear 16B chunks) ----
    {
        const bf16x8* src = (const bf16x8*)Uhp;
        bf16x8*       dst = (bf16x8*)UhL;
#pragma unroll
        for (int e = 0; e < (R_ * I_ / 8) / 256; ++e)   // 16 iters
            dst[tid + e * 256] = src[tid + e * 256];
    }

    // ---- stage h0 into LDS (bf16) ----
#pragma unroll
    for (int e = 0; e < 32; ++e) {
        int idx = tid + e * 256;
        int m = idx >> 9, r = idx & (R_ - 1);
        Xbf[m][r] = (__bf16)h0[(size_t)(b0 + m) * R_ + r];
    }
    __syncthreads();

    const size_t ysz = (size_t)B_ * T_ * O_;

    for (int t = 0; t < T_; ++t) {
        // Opaque zero offsets: compiler must treat streamed-weight addresses
        // as loop-varying -> loads stay inside the loop (hit hot L2/LDS each
        // step) instead of being hoisted, held in 700 VGPRs, and spilled.
        long          wofs = 0;   // for global packed W_h / W_out (SGPR)
        int           uofs = 0;   // for LDS packed U_h (VGPR)
        asm volatile("" : "+s"(wofs));
        asm volatile("" : "+v"(uofs));
        const __bf16* WhS = Whp + wofs;
        const __bf16* WoS = Woutp + wofs;
        const __bf16* UhS = UhL + uofs;

        // ---- stage x_pred[b0..b0+15, t, :] -> bf16 LDS ----
#pragma unroll
        for (int e = 0; e < 4; ++e) {
            int idx = tid + e * 256;
            int m = idx >> 6, i = idx & 63;
            XPbf[m][i] = (__bf16)x_pred[(size_t)(b0 + m) * T_ * I_ + (size_t)t * I_ + i];
        }
        __syncthreads();                           // B1: XP ready, Xbf stable

        f32x8 acc[4] = {};

        // hidden matmul: VGPR-resident K quarter
#pragma unroll
        for (int kc = 0; kc < 4; ++kc) {
            bf16x16 a = load_afrag(&Xbf[0][0], LDX, lm, kc * 32 + 8 * half);
#pragma unroll
            for (int nt = 0; nt < 4; ++nt)
                acc[nt] = WMMA_BF16(a, BWh[nt][kc], acc[nt]);
        }
        // hidden matmul: L2-streamed remainder (packed, coalesced)
#pragma unroll
        for (int kc = 4; kc < 16; ++kc) {
            bf16x16 a = load_afrag(&Xbf[0][0], LDX, lm, kc * 32 + 8 * half);
#pragma unroll
            for (int nt = 0; nt < 4; ++nt) {
                bf16x16 b = ldfrag(WhS, wave * 4 + nt, 16, kc, lane);
                acc[nt] = WMMA_BF16(a, b, acc[nt]);
            }
        }
        // + u_t = x_pred_tile @ U_h^T, U_h frags from LDS
#pragma unroll
        for (int kc = 0; kc < 2; ++kc) {
            bf16x16 a = load_afrag(&XPbf[0][0], LDP, lm, kc * 32 + 8 * half);
#pragma unroll
            for (int nt = 0; nt < 4; ++nt) {
                bf16x16 b = ldfrag(UhS, wave * 4 + nt, 2, kc, lane);
                acc[nt] = WMMA_BF16(a, b, acc[nt]);
            }
        }
        __syncthreads();                           // B2: all reads of old Xbf done

        // ---- tanh (fp32) + write new hidden state ----
#pragma unroll
        for (int nt = 0; nt < 4; ++nt) {
            const int n = wave * 64 + nt * 16 + lm;
            const float bh = bhv[nt];
#pragma unroll
            for (int v = 0; v < 8; ++v) {
                const int m = v + 8 * half;        // C/D layout: M = v + 8*half
                float val = tanhf(acc[nt][v] + bh);
                Xbf[m][n] = (__bf16)val;
                if (t == T_ - 1) {                 // x_last (tuple repeats it twice)
                    out[ysz + (size_t)(b0 + m) * R_ + n] = val;
                    out[ysz + (size_t)B_ * R_ + (size_t)(b0 + m) * R_ + n] = val;
                }
            }
        }
        __syncthreads();                           // B3: new Xbf ready

        // ---- y_t = x @ W_out^T + b_out : waves 0-3, one 16-col O-tile each ----
        if (wave < 4) {
            f32x8 yacc = {};
#pragma unroll
            for (int kc = 0; kc < 16; ++kc) {
                bf16x16 a = load_afrag(&Xbf[0][0], LDX, lm, kc * 32 + 8 * half);
                bf16x16 b = ldfrag(WoS, wave, 16, kc, lane);
                yacc = WMMA_BF16(a, b, yacc);
            }
            const int o = wave * 16 + lm;
#pragma unroll
            for (int v = 0; v < 8; ++v) {
                const int m = v + 8 * half;
                out[(size_t)(b0 + m) * T_ * O_ + (size_t)t * O_ + o] = yacc[v] + bov;
            }
        }
        // next iteration's B1 protects Xbf against early overwrite
    }
}

extern "C" void kernel_launch(void* const* d_in, const int* in_sizes, int n_in,
                              void* d_out, int out_size, void* d_ws, size_t ws_size,
                              hipStream_t stream) {
    const float* x_pred = (const float*)d_in[0];
    const float* h0     = (const float*)d_in[1];
    const float* W_h    = (const float*)d_in[2];
    const float* b_h    = (const float*)d_in[3];
    const float* U_h    = (const float*)d_in[4];
    const float* W_out  = (const float*)d_in[5];
    const float* b_out  = (const float*)d_in[6];
    float* out = (float*)d_out;
    __bf16* wbf = (__bf16*)d_ws;   // needs (R*R + R*I + O*R)*2 = 640 KB

    const int total = R_ * R_ + R_ * I_ + O_ * R_;           // 327680
    pack_weights_kernel<<<(total + 255) / 256, 256, 0, stream>>>(W_h, U_h, W_out, wbf);
    rnn_scan_kernel<<<B_ / 16, 256, 0, stream>>>(x_pred, h0, b_h, b_out, wbf, out);
}
